// CustomTransformerLayer_46334107189391
// MI455X (gfx1250) — matrix operand order repaired and checked by reading, not compiled, not run on hardware
//
#include <hip/hip_runtime.h>
#include <hip/hip_bf16.h>

typedef _Float16 v16h __attribute__((ext_vector_type(16)));
typedef _Float16 v8h  __attribute__((ext_vector_type(8)));
typedef float    v8f  __attribute__((ext_vector_type(8)));
typedef unsigned int u32x4 __attribute__((ext_vector_type(4)));
typedef int          i32x8 __attribute__((ext_vector_type(8)));
typedef int          i32x4 __attribute__((ext_vector_type(4)));

#define EMBED 1024
#define DFF   4096
#define BATCH 4
#define SEQ   2048
#define TOK   (BATCH * SEQ)   // 8192 rows total

// ------------------------------------------------------------- f32 -> f16
__global__ __launch_bounds__(256) void cvt_f32_f16(const float* __restrict__ in,
                                                   _Float16* __restrict__ out, int n) {
  int i = blockIdx.x * blockDim.x + threadIdx.x;
  int stride = gridDim.x * blockDim.x;
  for (; i < n; i += stride) out[i] = (_Float16)in[i];
}

// ------------------------------------- f32 [K,N] -> f16 transposed [N,K]
__global__ __launch_bounds__(256) void cvt_f32_f16_t(const float* __restrict__ in,
                                                     _Float16* __restrict__ out,
                                                     int Kd, int Nd) {
  __shared__ float t[32][33];
  const int bn = blockIdx.x * 32;
  const int bk = blockIdx.y * 32;
  const int tx = threadIdx.x & 31, ty = threadIdx.x >> 5;  // 32 x 8
#pragma unroll
  for (int e = 0; e < 32; e += 8)
    t[ty + e][tx] = in[(long long)(bk + ty + e) * Nd + (bn + tx)];
  __syncthreads();
#pragma unroll
  for (int e = 0; e < 32; e += 8)
    out[(long long)(bn + ty + e) * Kd + (bk + tx)] = (_Float16)t[tx][ty + e];
}

// ------------------------------------------------------------- TDM helper
// 2D tile load: 128 rows x 32 f16 (64B) into LDS, per ISA D# groups (§8.3/8.4).
__device__ __forceinline__ void tdm_load_tile_f16(const _Float16* gtile, unsigned lds_off,
                                                  unsigned width_elems, unsigned rows,
                                                  unsigned stride_elems) {
  unsigned long long ga = (unsigned long long)gtile;
  u32x4 g0;
  g0[0] = 1u;                                           // count=1, user descriptor
  g0[1] = lds_off;                                      // lds_addr (bytes)
  g0[2] = (unsigned)ga;                                 // global_addr[31:0]
  g0[3] = (unsigned)((ga >> 32) & 0x1ffffffULL)         // global_addr[56:32]
          | (2u << 30);                                 // type = 2 ("image")
  i32x8 g1;
  g1[0] = 1 << 16;                                      // data_size=1 (2 bytes)
  g1[1] = (int)(width_elems << 16);                     // tensor_dim0[15:0]
  g1[2] = (int)((width_elems >> 16) | (rows << 16));    // dim0[31:16] | dim1[15:0]
  g1[3] = (int)((rows >> 16) | (32u << 16));            // dim1[31:16] | tile_dim0=32
  g1[4] = 128;                                          // tile_dim1=128, tile_dim2=0
  g1[5] = (int)stride_elems;                            // tensor_dim0_stride[31:0]
  g1[6] = 0;                                            // stride hi | dim1_stride lo
  g1[7] = 0;
  i32x4 gz4;
  gz4[0] = gz4[1] = gz4[2] = gz4[3] = 0;                // 2D: groups 2/3 unused
  i32x8 gz8;
  gz8[0] = gz8[1] = gz8[2] = gz8[3] = 0;
  gz8[4] = gz8[5] = gz8[6] = gz8[7] = 0;
  __builtin_amdgcn_tensor_load_to_lds(g0, g1, gz4, gz4, gz8, 0);
}

// ------------------------------------------------------------- WMMA GEMM
// C[M,N] = A[M,K] x B^T + bias ; A row-major [M,K] f16, B row-major [N,K] f16.
// Block = 256 threads (8 wave32s), tile 128x128, K-step 32, TDM double-buffered
// LDS staging (wave 0 drives the DMA), 8 v_wmma_f32_16x16x32_f16 per wave/K-step.
template <bool RELU, bool OUTF16, bool TRANSC>
__global__ __launch_bounds__(256)
void wmma_gemm(const _Float16* __restrict__ A, int lda, long long sA,
               const _Float16* __restrict__ B, int ldb, long long sB,
               const float* __restrict__ bias,
               void* __restrict__ Cv, int ldc, long long sC, int K) {
  A += (long long)blockIdx.z * sA;
  B += (long long)blockIdx.z * sB;
  const int bm0 = blockIdx.y * 128;
  const int bn0 = blockIdx.x * 128;
  const int tid  = threadIdx.x;
  const int lane = tid & 31;
  const int wave = tid >> 5;     // 0..7
  const int wm   = wave >> 1;    // 0..3 (M)
  const int wn   = wave & 1;     // 0..1 (N)
  const int lh   = lane >> 4;
  const int l15  = lane & 15;

  __shared__ __align__(16) _Float16 As[2][128][32];   // [buf][m][k]
  __shared__ __align__(16) _Float16 Bs[2][128][32];   // [buf][n][k]

  const unsigned rowsA = gridDim.y * 128u;
  const unsigned rowsB = gridDim.x * 128u;
  const _Float16* Abase = A + (long long)bm0 * lda;
  const _Float16* Bbase = B + (long long)bn0 * ldb;
  const unsigned ldsA0 = (unsigned)(unsigned long long)&As[0][0][0];
  const unsigned ldsA1 = (unsigned)(unsigned long long)&As[1][0][0];
  const unsigned ldsB0 = (unsigned)(unsigned long long)&Bs[0][0][0];
  const unsigned ldsB1 = (unsigned)(unsigned long long)&Bs[1][0][0];

  v8f acc[2][4];
#pragma unroll
  for (int i = 0; i < 2; ++i)
#pragma unroll
    for (int j = 0; j < 4; ++j)
#pragma unroll
      for (int r = 0; r < 8; ++r) acc[i][j][r] = 0.0f;

  // prime buffer 0
  if (wave == 0) {
    tdm_load_tile_f16(Abase, ldsA0, (unsigned)K, rowsA, (unsigned)lda);
    tdm_load_tile_f16(Bbase, ldsB0, (unsigned)K, rowsB, (unsigned)ldb);
  }

  int buf = 0;
  for (int k0 = 0; k0 < K; k0 += 32, buf ^= 1) {
    if (wave == 0) {
      if (k0 + 32 < K) {  // kick next slab's DMA, then wait for current pair
        tdm_load_tile_f16(Abase + (k0 + 32), buf ? ldsA0 : ldsA1,
                          (unsigned)K, rowsA, (unsigned)lda);
        tdm_load_tile_f16(Bbase + (k0 + 32), buf ? ldsB0 : ldsB1,
                          (unsigned)K, rowsB, (unsigned)ldb);
        __builtin_amdgcn_s_wait_tensorcnt(2);
      } else {
        __builtin_amdgcn_s_wait_tensorcnt(0);
      }
    }
    __syncthreads();  // current buffer ready for all waves

    v16h af[2], bf[4];
#pragma unroll
    for (int i = 0; i < 2; ++i) {
      int r = wm * 32 + i * 16 + l15;                    // A row (M)
      v8h lo = *(const v8h*)&As[buf][r][lh * 8];         // K = 8*lh + 0..7
      v8h hi = *(const v8h*)&As[buf][r][16 + lh * 8];    // K = 16 + 8*lh + 0..7
      af[i] = __builtin_shufflevector(lo, hi, 0,1,2,3,4,5,6,7,8,9,10,11,12,13,14,15);
    }
#pragma unroll
    for (int j = 0; j < 4; ++j) {
      int n = wn * 64 + j * 16 + l15;                    // B col (N)
      v8h lo = *(const v8h*)&Bs[buf][n][lh * 8];         // K = 8*lh + 0..7
      v8h hi = *(const v8h*)&Bs[buf][n][16 + lh * 8];    // K = 16 + 8*lh + 0..7
      bf[j] = __builtin_shufflevector(lo, hi, 0,1,2,3,4,5,6,7,8,9,10,11,12,13,14,15);
    }
#pragma unroll
    for (int i = 0; i < 2; ++i)
#pragma unroll
      for (int j = 0; j < 4; ++j)
        acc[i][j] = __builtin_amdgcn_wmma_f32_16x16x32_f16(
            false, af[i], false, bf[j], (short)0, acc[i][j], false, false);

    __syncthreads();  // all reads of `buf` done before its next DMA overwrite
  }

  // epilogue: C/D layout: VGPR r -> M = r + 8*lh, lane&15 -> N
#pragma unroll
  for (int j = 0; j < 4; ++j) {
    int n = bn0 + wn * 64 + j * 16 + l15;
    float bv = bias ? bias[n] : 0.0f;
#pragma unroll
    for (int i = 0; i < 2; ++i) {
#pragma unroll
      for (int r = 0; r < 8; ++r) {
        int m = bm0 + wm * 32 + i * 16 + lh * 8 + r;
        float v = acc[i][j][r] + bv;
        if (RELU) v = v > 0.0f ? v : 0.0f;
        long long off = TRANSC ? ((long long)n * ldc + m)
                               : ((long long)blockIdx.z * sC + (long long)m * ldc + n);
        if (OUTF16) ((_Float16*)Cv)[off] = (_Float16)v;
        else        ((float*)Cv)[off]    = v;
      }
    }
  }
}

// ------------------------------------------------------------- row softmax
__global__ __launch_bounds__(256)
void softmax_rows(const float* __restrict__ S, _Float16* __restrict__ P, int n) {
  const long long base = (long long)blockIdx.x * n;
  const int tid = threadIdx.x;
  __shared__ float red[256];
  float lmax = -3.4e38f;
  for (int i = tid; i < n; i += 256) lmax = fmaxf(lmax, S[base + i]);
  red[tid] = lmax; __syncthreads();
  for (int s = 128; s > 0; s >>= 1) { if (tid < s) red[tid] = fmaxf(red[tid], red[tid + s]); __syncthreads(); }
  const float m = red[0]; __syncthreads();
  float lsum = 0.0f;
  for (int i = tid; i < n; i += 256) lsum += __expf(S[base + i] - m);
  red[tid] = lsum; __syncthreads();
  for (int s = 128; s > 0; s >>= 1) { if (tid < s) red[tid] += red[tid + s]; __syncthreads(); }
  const float inv = 1.0f / red[0];
  for (int i = tid; i < n; i += 256) P[base + i] = (_Float16)(__expf(S[base + i] - m) * inv);
}

// ----------------------------------------------- residual add + LayerNorm
__global__ __launch_bounds__(256)
void add_ln(const float* __restrict__ a, const float* __restrict__ b,
            const float* __restrict__ g, const float* __restrict__ be,
            float* __restrict__ o32, _Float16* __restrict__ o16) {
  const long long base = (long long)blockIdx.x * EMBED;
  const int tid = threadIdx.x;
  __shared__ float r1[256], r2[256];
  float v[4];
  float s = 0.0f, ss = 0.0f;
#pragma unroll
  for (int e = 0; e < 4; ++e) {
    int i = tid + e * 256;
    float x = a[base + i] + b[base + i];
    v[e] = x; s += x; ss += x * x;
  }
  r1[tid] = s; r2[tid] = ss; __syncthreads();
  for (int st = 128; st > 0; st >>= 1) {
    if (tid < st) { r1[tid] += r1[tid + st]; r2[tid] += r2[tid + st]; }
    __syncthreads();
  }
  const float mean = r1[0] * (1.0f / EMBED);
  const float var  = r2[0] * (1.0f / EMBED) - mean * mean;
  const float rstd = rsqrtf(var + 1e-5f);
#pragma unroll
  for (int e = 0; e < 4; ++e) {
    int i = tid + e * 256;
    float y = (v[e] - mean) * rstd * g[i] + be[i];
    if (o32) o32[base + i] = y;
    if (o16) o16[base + i] = (_Float16)y;
  }
}

// ------------------------------------------------------------- launcher
extern "C" void kernel_launch(void* const* d_in, const int* in_sizes, int n_in,
                              void* d_out, int out_size, void* d_ws, size_t ws_size,
                              hipStream_t stream) {
  const float* src = (const float*)d_in[0];
  const float* Wq  = (const float*)d_in[1];  const float* bq  = (const float*)d_in[2];
  const float* Wk  = (const float*)d_in[3];  const float* bk  = (const float*)d_in[4];
  const float* Wv  = (const float*)d_in[5];  const float* bv  = (const float*)d_in[6];
  const float* Wo  = (const float*)d_in[7];  const float* bo  = (const float*)d_in[8];
  const float* W1  = (const float*)d_in[9];  const float* b1  = (const float*)d_in[10];
  const float* W2  = (const float*)d_in[11]; const float* b2  = (const float*)d_in[12];
  const float* g1  = (const float*)d_in[13]; const float* be1 = (const float*)d_in[14];
  const float* g2  = (const float*)d_in[15]; const float* be2 = (const float*)d_in[16];
  (void)in_sizes; (void)n_in; (void)out_size; (void)ws_size;

  char* ws = (char*)d_ws;
  size_t off = 0;
  auto carve = [&](size_t bytes) -> char* {
    char* p = ws + off;
    off = (off + bytes + 255) & ~(size_t)255;
    return p;
  };

  _Float16* srch = (_Float16*)carve((size_t)TOK * EMBED * 2);
  _Float16* Wqt  = (_Float16*)carve((size_t)EMBED * EMBED * 2);  // [N,K]
  _Float16* Wkt  = (_Float16*)carve((size_t)EMBED * EMBED * 2);
  _Float16* Wvt  = (_Float16*)carve((size_t)EMBED * EMBED * 2);
  _Float16* Wot  = (_Float16*)carve((size_t)EMBED * EMBED * 2);
  _Float16* W1t  = (_Float16*)carve((size_t)DFF * EMBED * 2);    // [DFF,EMBED]
  _Float16* W2t  = (_Float16*)carve((size_t)EMBED * DFF * 2);    // [EMBED,DFF]
  _Float16* Qh   = (_Float16*)carve((size_t)TOK * EMBED * 2);
  _Float16* Kh   = (_Float16*)carve((size_t)TOK * EMBED * 2);
  _Float16* VhT  = (_Float16*)carve((size_t)EMBED * TOK * 2);    // [EMBED][TOK]
  float*    S    = (float*)   carve((size_t)BATCH * SEQ * SEQ * 4); // reused: Y, FF
  _Float16* P    = (_Float16*)carve((size_t)BATCH * SEQ * SEQ * 2);
  _Float16* AOh  = (_Float16*)carve((size_t)TOK * EMBED * 2);
  float*    X1   = (float*)   carve((size_t)TOK * EMBED * 4);
  _Float16* X1h  = (_Float16*)carve((size_t)TOK * EMBED * 2);
  _Float16* H1h  = (_Float16*)carve((size_t)TOK * DFF * 2);

  const dim3 blk(256);
  cvt_f32_f16<<<dim3((TOK * EMBED + 1023) / 1024), blk, 0, stream>>>(src, srch, TOK * EMBED);
  cvt_f32_f16_t<<<dim3(EMBED / 32, EMBED / 32), blk, 0, stream>>>(Wq, Wqt, EMBED, EMBED);
  cvt_f32_f16_t<<<dim3(EMBED / 32, EMBED / 32), blk, 0, stream>>>(Wk, Wkt, EMBED, EMBED);
  cvt_f32_f16_t<<<dim3(EMBED / 32, EMBED / 32), blk, 0, stream>>>(Wv, Wvt, EMBED, EMBED);
  cvt_f32_f16_t<<<dim3(EMBED / 32, EMBED / 32), blk, 0, stream>>>(Wo, Wot, EMBED, EMBED);
  cvt_f32_f16_t<<<dim3(DFF / 32, EMBED / 32), blk, 0, stream>>>(W1, W1t, EMBED, DFF);
  cvt_f32_f16_t<<<dim3(EMBED / 32, DFF / 32), blk, 0, stream>>>(W2, W2t, DFF, EMBED);

  const long long sQKV = (long long)SEQ * EMBED;
  const long long sSc  = (long long)SEQ * SEQ;

  // Q/K projections -> f16 [TOK,EMBED]; V projection -> f16 transposed [EMBED,TOK]
  wmma_gemm<false,true,false><<<dim3(EMBED/128, TOK/128, 1), blk, 0, stream>>>(
      srch, EMBED, 0, Wqt, EMBED, 0, bq, Qh, EMBED, 0, EMBED);
  wmma_gemm<false,true,false><<<dim3(EMBED/128, TOK/128, 1), blk, 0, stream>>>(
      srch, EMBED, 0, Wkt, EMBED, 0, bk, Kh, EMBED, 0, EMBED);
  wmma_gemm<false,true,true><<<dim3(EMBED/128, TOK/128, 1), blk, 0, stream>>>(
      srch, EMBED, 0, Wvt, EMBED, 0, bv, VhT, TOK, 0, EMBED);

  // scores = Q @ K^T per batch (no 1/sqrt(d), per reference) -> f32
  wmma_gemm<false,false,false><<<dim3(SEQ/128, SEQ/128, BATCH), blk, 0, stream>>>(
      Qh, EMBED, sQKV, Kh, EMBED, sQKV, nullptr, S, SEQ, sSc, EMBED);

  softmax_rows<<<dim3(BATCH * SEQ), blk, 0, stream>>>(S, P, SEQ);

  // attn @ V per batch: B = VhT[:, z*SEQ + k], ldb = TOK -> f16
  wmma_gemm<false,true,false><<<dim3(EMBED/128, SEQ/128, BATCH), blk, 0, stream>>>(
      P, SEQ, sSc, VhT, TOK, (long long)SEQ, nullptr, AOh, EMBED, sQKV, SEQ);

  // output projection -> f32 (into S region, now free)
  wmma_gemm<false,false,false><<<dim3(EMBED/128, TOK/128, 1), blk, 0, stream>>>(
      AOh, EMBED, 0, Wot, EMBED, 0, bo, S, EMBED, 0, EMBED);

  // x = LN(src + attn_out)
  add_ln<<<dim3(TOK), blk, 0, stream>>>(src, S, g1, be1, X1, X1h);

  // ff1 = relu(x @ W1 + b1) -> f16
  wmma_gemm<true,true,false><<<dim3(DFF/128, TOK/128, 1), blk, 0, stream>>>(
      X1h, EMBED, 0, W1t, EMBED, 0, b1, H1h, DFF, 0, EMBED);

  // ff2 = h1 @ W2 + b2 -> f32 (into S region, free again)
  wmma_gemm<false,false,false><<<dim3(EMBED/128, TOK/128, 1), blk, 0, stream>>>(
      H1h, DFF, 0, W2t, DFF, 0, b2, S, EMBED, 0, DFF);

  // out = LN(x + ff)
  add_ln<<<dim3(TOK), blk, 0, stream>>>(X1, S, g2, be2, (float*)d_out, nullptr);
}
